// MultiHeadAttention_3521873183085
// MI455X (gfx1250) — matrix operand order repaired and checked by reading, not compiled, not run on hardware
//
#include <hip/hip_runtime.h>
#include <hip/hip_bf16.h>

typedef __bf16 v16bf __attribute__((ext_vector_type(16)));
typedef __bf16 v8bf  __attribute__((ext_vector_type(8)));
typedef __bf16 v4bf  __attribute__((ext_vector_type(4)));
typedef float  v8f   __attribute__((ext_vector_type(8)));

namespace {
constexpr int kB  = 4;
constexpr int kS  = 2048;
constexpr int kD  = 1024;
constexpr int kH  = 16;
constexpr int kDH = 64;
constexpr int k3D = 3 * kD;
constexpr float kScale = 0.125f;   // 1/sqrt(64)
}

// ---------------------------------------------------------------------------
// fp32 -> bf16 conversion (4 elements / thread)
// ---------------------------------------------------------------------------
__global__ __launch_bounds__(256) void cvt_bf16(const float* __restrict__ in,
                                                __bf16* __restrict__ out, int n4) {
  int i = blockIdx.x * 256 + threadIdx.x;
  if (i >= n4) return;
  const float4 v = ((const float4*)in)[i];
  v4bf o;
  o[0] = (__bf16)v.x; o[1] = (__bf16)v.y; o[2] = (__bf16)v.z; o[3] = (__bf16)v.w;
  *(v4bf*)(out + (size_t)i * 4) = o;
}

// ---------------------------------------------------------------------------
// QKV projection: [B*S, D] x [D, 3D] + bias -> Q/K/V bf16 [B,H,S,DH]
// One wave computes a 16x64 tile (4 WMMA accumulators), K-loop step 32.
// ---------------------------------------------------------------------------
__global__ __launch_bounds__(256) void qkv_gemm(const __bf16* __restrict__ xb,
                                                const __bf16* __restrict__ wb,
                                                const float* __restrict__ bias,
                                                __bf16* __restrict__ qo,
                                                __bf16* __restrict__ ko,
                                                __bf16* __restrict__ vo) {
  const int lane = threadIdx.x & 31;
  const int wid  = threadIdx.x >> 5;
  const int tile = blockIdx.x * 8 + wid;
  const int nTN  = k3D / 64;               // 48
  const int tm   = tile / nTN;
  const int tn   = tile % nTN;
  const int row0 = tm * 16;
  const int col0 = tn * 64;
  const int m  = lane & 15;
  const int kh = lane >> 4;

  v8f acc[4];
#pragma unroll
  for (int t = 0; t < 4; ++t)
#pragma unroll
    for (int r = 0; r < 8; ++r) acc[t][r] = 0.f;

  for (int kk = 0; kk < kD; kk += 32) {
    // A operand: row m of x, K split per ISA 16-bit A layout
    const __bf16* ap = xb + (size_t)(row0 + m) * kD + kk + kh * 8;
    v8bf alo = *(const v8bf*)ap;
    v8bf ahi = *(const v8bf*)(ap + 16);
    v16bf a;
#pragma unroll
    for (int i = 0; i < 8; ++i) { a[i] = alo[i]; a[i + 8] = ahi[i]; }

    // B operand: lane holds column m, K rows in table order (kk + kb + i, kk + kb + 16 + i)
    const int kb = 8 * kh;
    const __bf16* bp = wb + (size_t)(kk + kb) * k3D + col0 + m;
    if (kk + 32 < kD)
      __builtin_prefetch(bp + (size_t)32 * k3D, 0, 3);
#pragma unroll
    for (int t = 0; t < 4; ++t) {
      v16bf bv;
#pragma unroll
      for (int i = 0; i < 8; ++i) { bv[i] = bp[(size_t)i * k3D + t * 16]; bv[i + 8] = bp[(size_t)(16 + i) * k3D + t * 16]; }
      acc[t] = __builtin_amdgcn_wmma_f32_16x16x32_bf16(false, a, false, bv,
                                                       (short)0, acc[t], false, false);
    }
  }

  // Epilogue: bias + scatter into [B,H,S,DH]
#pragma unroll
  for (int t = 0; t < 4; ++t) {
#pragma unroll
    for (int r = 0; r < 8; ++r) {
      const int c   = col0 + t * 16 + m;
      const int row = row0 + r + 8 * kh;
      const float val = acc[t][r] + bias[c];
      const int mtx = c >> 10;          // 0=Q 1=K 2=V
      const int dd  = c & 1023;
      const int h   = dd >> 6;
      const int dh  = dd & 63;
      const int bb  = row >> 11;        // / S
      const int ss  = row & 2047;
      __bf16* dst = (mtx == 0) ? qo : ((mtx == 1) ? ko : vo);
      dst[(((size_t)bb * kH + h) * kS + ss) * kDH + dh] = (__bf16)val;
    }
  }
}

// ---------------------------------------------------------------------------
// Flash attention, 64 keys per block-iteration, double-buffered staging.
//   block = 4 waves; K staged *transposed* (KT[dh][key]) via ds stores,
//   V staged row-major via global_load_async_to_lds_b128 (ASYNCcnt),
//   next block's V issued async + K loaded to regs while current block's
//   16 WMMAs + online softmax execute. 40 KB LDS of the 320 KB WGP pool.
// ---------------------------------------------------------------------------
__global__ __launch_bounds__(128) void attn_kernel(const __bf16* __restrict__ qb,
                                                   const __bf16* __restrict__ kb,
                                                   const __bf16* __restrict__ vb,
                                                   __bf16* __restrict__ attnb) {
  __shared__ __align__(32) __bf16 KT[2][kDH * 64];   // [buf][dh][key]  16 KB
  __shared__ __align__(32) __bf16 Vt[2][64 * kDH];   // [buf][key][dh]  16 KB
  __shared__ __align__(32) __bf16 Pt[4][16 * 64];    // per-wave P       8 KB

  const int tid  = threadIdx.x;
  const int lane = tid & 31;
  const int wid  = tid >> 5;
  const int m    = lane & 15;
  const int kh   = lane >> 4;

  const int qc = blockIdx.x & 31;          // S/64 q-chunks
  const int bh = blockIdx.x >> 5;          // b*H + h
  const int b  = bh >> 4;
  const int h  = bh & 15;

  const __bf16* Qp = qb + (size_t)bh * kS * kDH;
  const __bf16* Kp = kb + (size_t)bh * kS * kDH;
  const __bf16* Vp = vb + (size_t)bh * kS * kDH;

  const int q0 = qc * 64 + wid * 16;

  // Q tile as two A operands (DH = 64 = 2 x K32)
  v16bf qa[2];
#pragma unroll
  for (int c = 0; c < 2; ++c) {
    const __bf16* qp = Qp + (size_t)(q0 + m) * kDH + c * 32 + kh * 8;
    v8bf lo = *(const v8bf*)qp;
    v8bf hi = *(const v8bf*)(qp + 16);
#pragma unroll
    for (int i = 0; i < 8; ++i) { qa[c][i] = lo[i]; qa[c][i + 8] = hi[i]; }
  }

  v8f oacc[4];
  float rm[8], rl[8];
#pragma unroll
  for (int r = 0; r < 8; ++r) { rm[r] = -3.0e38f; rl[r] = 0.f; }
#pragma unroll
  for (int t = 0; t < 4; ++t)
#pragma unroll
    for (int r = 0; r < 8; ++r) oacc[t][r] = 0.f;

  // staging assignment: 128 threads cover 64 keys x 64 dh (64 B each/tile)
  const int krow = tid >> 1;               // 0..63
  const int dh0  = (tid & 1) * 32;         // 0 or 32
  constexpr int nJ = kS / 64;              // 32

  // ---- prologue: stage block 0 into buffer 0 ----
  {
    const unsigned vlds = (unsigned)(unsigned long long)&Vt[0][krow * kDH + dh0];
    const unsigned long long vga =
        (unsigned long long)(Vp + (size_t)krow * kDH + dh0);
    asm volatile("global_load_async_to_lds_b128 %0, %1, off"            :: "v"(vlds), "v"(vga) : "memory");
    asm volatile("global_load_async_to_lds_b128 %0, %1, off offset:16"  :: "v"(vlds), "v"(vga) : "memory");
    asm volatile("global_load_async_to_lds_b128 %0, %1, off offset:32"  :: "v"(vlds), "v"(vga) : "memory");
    asm volatile("global_load_async_to_lds_b128 %0, %1, off offset:48"  :: "v"(vlds), "v"(vga) : "memory");
    const __bf16* kp = Kp + (size_t)krow * kDH + dh0;
#pragma unroll
    for (int c2 = 0; c2 < 4; ++c2) {
      v8bf kvv = *(const v8bf*)(kp + c2 * 8);
#pragma unroll
      for (int i = 0; i < 8; ++i)
        KT[0][(dh0 + c2 * 8 + i) * 64 + krow] = kvv[i];
    }
  }

  for (int j = 0; j < nJ; ++j) {
    const int cur = j & 1;
    const int nxt = cur ^ 1;
    asm volatile("s_wait_asynccnt 0x0" ::: "memory");
    __syncthreads();

    // ---- issue next block's staging (overlaps this block's compute) ----
    v8bf knext[4];
    const bool havenext = (j + 1 < nJ);
    if (havenext) {
      const unsigned vlds = (unsigned)(unsigned long long)&Vt[nxt][krow * kDH + dh0];
      const unsigned long long vga =
          (unsigned long long)(Vp + (size_t)((j + 1) * 64 + krow) * kDH + dh0);
      asm volatile("global_load_async_to_lds_b128 %0, %1, off"            :: "v"(vlds), "v"(vga) : "memory");
      asm volatile("global_load_async_to_lds_b128 %0, %1, off offset:16"  :: "v"(vlds), "v"(vga) : "memory");
      asm volatile("global_load_async_to_lds_b128 %0, %1, off offset:32"  :: "v"(vlds), "v"(vga) : "memory");
      asm volatile("global_load_async_to_lds_b128 %0, %1, off offset:48"  :: "v"(vlds), "v"(vga) : "memory");
      const __bf16* kp = Kp + (size_t)((j + 1) * 64 + krow) * kDH + dh0;
#pragma unroll
      for (int c2 = 0; c2 < 4; ++c2) knext[c2] = *(const v8bf*)(kp + c2 * 8);
    }

    // ---- scores = Q * K^T : 4 N-tiles (16 keys each) x 2 K-chunks ----
    v8f sc[4];
#pragma unroll
    for (int nt = 0; nt < 4; ++nt)
#pragma unroll
      for (int r = 0; r < 8; ++r) sc[nt][r] = 0.f;
    const int kd = 8 * kh;                 // B-operand reduction row base (table order)
#pragma unroll
    for (int c = 0; c < 2; ++c) {
#pragma unroll
      for (int nt = 0; nt < 4; ++nt) {
        v16bf bbv;
#pragma unroll
        for (int i = 0; i < 8; ++i) { bbv[i] = KT[cur][(c * 32 + kd + i) * 64 + nt * 16 + m]; bbv[i + 8] = KT[cur][(c * 32 + kd + 16 + i) * 64 + nt * 16 + m]; }
        sc[nt] = __builtin_amdgcn_wmma_f32_16x16x32_bf16(false, qa[c], false, bbv,
                                                         (short)0, sc[nt], false, false);
      }
    }

    // ---- online softmax over 64 key columns (one update per 64 keys) ----
#pragma unroll
    for (int r = 0; r < 8; ++r) {
      const float s0 = sc[0][r] * kScale;
      const float s1 = sc[1][r] * kScale;
      const float s2 = sc[2][r] * kScale;
      const float s3 = sc[3][r] * kScale;
      float mx = fmaxf(fmaxf(s0, s1), fmaxf(s2, s3));
      mx = fmaxf(mx, __shfl_xor(mx, 1, 16));
      mx = fmaxf(mx, __shfl_xor(mx, 2, 16));
      mx = fmaxf(mx, __shfl_xor(mx, 4, 16));
      mx = fmaxf(mx, __shfl_xor(mx, 8, 16));
      const float mn = fmaxf(rm[r], mx);
      const float al = __expf(rm[r] - mn);
      const float e0 = __expf(s0 - mn);
      const float e1 = __expf(s1 - mn);
      const float e2 = __expf(s2 - mn);
      const float e3 = __expf(s3 - mn);
      float ps = (e0 + e1) + (e2 + e3);
      ps += __shfl_xor(ps, 1, 16);
      ps += __shfl_xor(ps, 2, 16);
      ps += __shfl_xor(ps, 4, 16);
      ps += __shfl_xor(ps, 8, 16);
      rl[r] = rl[r] * al + ps;
      rm[r] = mn;
      __bf16* prow = &Pt[wid][(r + 8 * kh) * 64 + m];
      prow[0]  = (__bf16)e0;
      prow[16] = (__bf16)e1;
      prow[32] = (__bf16)e2;
      prow[48] = (__bf16)e3;
#pragma unroll
      for (int t = 0; t < 4; ++t) oacc[t][r] *= al;
    }
    asm volatile("s_wait_dscnt 0x0" ::: "memory");

    // ---- PV: 2 K-chunks (32 keys) x 4 N-tiles over DH=64 ----
#pragma unroll
    for (int g = 0; g < 2; ++g) {
      v8bf plo = *(const v8bf*)&Pt[wid][m * 64 + g * 32 + kh * 8];
      v8bf phi = *(const v8bf*)&Pt[wid][m * 64 + g * 32 + 16 + kh * 8];
      v16bf pa;
#pragma unroll
      for (int i = 0; i < 8; ++i) { pa[i] = plo[i]; pa[i + 8] = phi[i]; }
      const int kv = g * 32 + 8 * kh;        // key row base within 64 (table order)
#pragma unroll
      for (int t = 0; t < 4; ++t) {
        v16bf vv;
#pragma unroll
        for (int i = 0; i < 8; ++i) { vv[i] = Vt[cur][(kv + i) * kDH + t * 16 + m]; vv[i + 8] = Vt[cur][(kv + 16 + i) * kDH + t * 16 + m]; }
        oacc[t] = __builtin_amdgcn_wmma_f32_16x16x32_bf16(false, pa, false, vv,
                                                          (short)0, oacc[t], false, false);
      }
    }

    // ---- scatter next K block transposed into the other buffer ----
    if (havenext) {
#pragma unroll
      for (int c2 = 0; c2 < 4; ++c2)
#pragma unroll
        for (int i = 0; i < 8; ++i)
          KT[nxt][(dh0 + c2 * 8 + i) * 64 + krow] = knext[c2][i];
    }
  }

  // normalize and write attn (bf16) back in [B,S,D]
#pragma unroll
  for (int t = 0; t < 4; ++t)
#pragma unroll
    for (int r = 0; r < 8; ++r) {
      const int s = q0 + r + 8 * kh;
      const float val = oacc[t][r] / rl[r];
      attnb[((size_t)b * kS + s) * kD + h * kDH + t * 16 + m] = (__bf16)val;
    }
}

// ---------------------------------------------------------------------------
// Output projection: attn [B*S, D] x w_out [D, D] + bias -> fp32 out
// ---------------------------------------------------------------------------
__global__ __launch_bounds__(256) void out_gemm(const __bf16* __restrict__ ab,
                                                const __bf16* __restrict__ wb,
                                                const float* __restrict__ bias,
                                                float* __restrict__ out) {
  const int lane = threadIdx.x & 31;
  const int wid  = threadIdx.x >> 5;
  const int tile = blockIdx.x * 8 + wid;
  const int nTN  = kD / 64;                 // 16
  const int tm   = tile / nTN;
  const int tn   = tile % nTN;
  const int row0 = tm * 16;
  const int col0 = tn * 64;
  const int m  = lane & 15;
  const int kh = lane >> 4;

  v8f acc[4];
#pragma unroll
  for (int t = 0; t < 4; ++t)
#pragma unroll
    for (int r = 0; r < 8; ++r) acc[t][r] = 0.f;

  for (int kk = 0; kk < kD; kk += 32) {
    const __bf16* ap = ab + (size_t)(row0 + m) * kD + kk + kh * 8;
    v8bf alo = *(const v8bf*)ap;
    v8bf ahi = *(const v8bf*)(ap + 16);
    v16bf a;
#pragma unroll
    for (int i = 0; i < 8; ++i) { a[i] = alo[i]; a[i + 8] = ahi[i]; }

    const int kb = 8 * kh;
    const __bf16* bp = wb + (size_t)(kk + kb) * kD + col0 + m;
    if (kk + 32 < kD)
      __builtin_prefetch(bp + (size_t)32 * kD, 0, 3);
#pragma unroll
    for (int t = 0; t < 4; ++t) {
      v16bf bv;
#pragma unroll
      for (int i = 0; i < 8; ++i) { bv[i] = bp[(size_t)i * kD + t * 16]; bv[i + 8] = bp[(size_t)(16 + i) * kD + t * 16]; }
      acc[t] = __builtin_amdgcn_wmma_f32_16x16x32_bf16(false, a, false, bv,
                                                       (short)0, acc[t], false, false);
    }
  }

#pragma unroll
  for (int t = 0; t < 4; ++t)
#pragma unroll
    for (int r = 0; r < 8; ++r) {
      const int c   = col0 + t * 16 + m;
      const int row = row0 + r + 8 * kh;
      out[(size_t)row * kD + c] = acc[t][r] + bias[c];
    }
}

// ---------------------------------------------------------------------------
extern "C" void kernel_launch(void* const* d_in, const int* in_sizes, int n_in,
                              void* d_out, int out_size, void* d_ws, size_t ws_size,
                              hipStream_t stream) {
  (void)in_sizes; (void)n_in; (void)out_size; (void)ws_size;
  const float* x     = (const float*)d_in[0];
  const float* w_qkv = (const float*)d_in[1];
  const float* b_qkv = (const float*)d_in[2];
  const float* w_out = (const float*)d_in[3];
  const float* b_out = (const float*)d_in[4];

  // workspace layout (bf16 elements)
  __bf16* p     = (__bf16*)d_ws;
  __bf16* xb    = p; p += (size_t)kB * kS * kD;        // 16 MB
  __bf16* wqkvb = p; p += (size_t)kD * k3D;            //  6 MB
  __bf16* woutb = p; p += (size_t)kD * kD;             //  2 MB
  __bf16* qbuf  = p; p += (size_t)kB * kH * kS * kDH;  // 16 MB
  __bf16* kbuf  = p; p += (size_t)kB * kH * kS * kDH;  // 16 MB
  __bf16* vbuf  = p; p += (size_t)kB * kH * kS * kDH;  // 16 MB
  __bf16* attnb = p;                                   // 16 MB

  // 1) fp32 -> bf16 conversions
  {
    const int n4x = kB * kS * kD / 4;
    cvt_bf16<<<n4x / 256, 256, 0, stream>>>(x, xb, n4x);
    const int n4w = kD * k3D / 4;
    cvt_bf16<<<n4w / 256, 256, 0, stream>>>(w_qkv, wqkvb, n4w);
    const int n4o = kD * kD / 4;
    cvt_bf16<<<n4o / 256, 256, 0, stream>>>(w_out, woutb, n4o);
  }

  // 2) QKV projection: 512 M-tiles * 48 N-tiles = 24576 waves / 8 per block
  qkv_gemm<<<3072, 256, 0, stream>>>(xb, wqkvb, b_qkv, qbuf, kbuf, vbuf);

  // 3) flash attention: B*H*(S/64) = 2048 blocks of 4 waves
  attn_kernel<<<2048, 128, 0, stream>>>(qbuf, kbuf, vbuf, attnb);

  // 4) output projection: 512 * 16 = 8192 waves / 8 per block
  out_gemm<<<1024, 256, 0, stream>>>(attnb, woutb, b_out, (float*)d_out);
}
